// LatentKVAttention_18356690223123
// MI455X (gfx1250) — matrix operand order repaired and checked by reading, not compiled, not run on hardware
//
#include <hip/hip_runtime.h>
#include <hip/hip_bf16.h>
#include <math.h>
#include <stdint.h>

typedef __attribute__((ext_vector_type(16))) _Float16 v16h;
typedef __attribute__((ext_vector_type(8)))  float    v8f;
typedef __attribute__((ext_vector_type(2)))  float    v2f;

#define Bn   2
#define Sn   1024
#define Dn   1024
#define Hn   16
#define HKVn 4
#define HDn  64
#define Nrow (Bn * Sn)      // 2048
#define KVD  (HKVn * HDn)   // 256
#define NCH  (Sn / 16)      // 64 chunks per (b,h)
#define NBH  (Bn * Hn)      // 32

// ---------------- WMMA helpers ----------------
static __device__ __forceinline__ v8f wmma_f16acc(v16h a, v16h b, v8f c) {
  return __builtin_amdgcn_wmma_f32_16x16x32_f16(false, a, false, b, (short)0, c, false, false);
}
static __device__ __forceinline__ v8f wmma_f32k4(v2f a, v2f b, v8f c) {
  return __builtin_amdgcn_wmma_f32_16x16x4_f32(false, a, false, b, (short)0, c, false, false);
}
static __device__ __forceinline__ void async_lds_b128(const void* g, unsigned ldsoff) {
  asm volatile("global_load_async_to_lds_b128 %0, %1, off"
               :: "v"(ldsoff), "v"((unsigned long long)(uintptr_t)g) : "memory");
}
static __device__ __forceinline__ void wait_async0() {
  asm volatile("s_wait_asynccnt 0" ::: "memory");
}
static __device__ __forceinline__ float featmap(float x) {
  return (x > 0.0f) ? (x + 1.0f) : __expf(x);   // elu(x)+1
}

// =====================================================================
// Weight pre-swizzle: W[K,M] (f32, row-major) -> f16 B-fragment stream.
// Fragment id = (kstep * (M/16) + coltile) * 32 + lane; 16 halves each.
// B frag element e of lane (n=lane&15, half=lane>>4):  k = e + 16*half.
// =====================================================================
__global__ __launch_bounds__(256)
void swizzle_w_kernel(const float* __restrict__ W, _Float16* __restrict__ Wf,
                      int Kk, int Mm)
{
  const int idx  = blockIdx.x * 256 + threadIdx.x;   // fragment id
  const int lane = idx & 31;
  const int t    = idx >> 5;
  const int mt16 = Mm >> 4;
  const int jt   = t % mt16;
  const int ks   = t / mt16;
  const int half = lane >> 4, ml = lane & 15;
  const float* src = W + (long)(ks * 32 + 16 * half) * Mm + jt * 16 + ml;
  v16h f;
  #pragma unroll
  for (int e = 0; e < 16; ++e) f[e] = (_Float16)src[(long)e * Mm];
  *(v16h*)(Wf + (long)idx * 16) = f;
}

// =====================================================================
// GEMM: C[N,M] = A[N,K] @ W[K,M] (+ bias); A fp32, W pre-swizzled f16.
// 128 threads = 4 independent waves; wave w -> C rows [64by+16w,+16) x 64 cols.
// LDS-free: A fragment in registers (4x b128 + pk-cvt); all 4 B fragments
// batch-loaded per k-step (8x b128 clause) then 4 WMMAs.
// =====================================================================
__global__ __launch_bounds__(128)
void gemm_wmma_kernel(const float* __restrict__ A, const _Float16* __restrict__ Bf,
                      const float* __restrict__ bias, float* __restrict__ C,
                      int Nn, int Kk, int Mm)
{
  const int tid  = threadIdx.x;
  const int wave = tid >> 5;
  const int lane = tid & 31;
  const int half = lane >> 4;
  const int ml   = lane & 15;
  const long tileN = (long)blockIdx.y * 64;
  const long tileM = (long)blockIdx.x * 64;
  const int  mt16  = Mm >> 4;

  v8f acc[4] = {};
  const float* arow = A + (tileN + wave * 16 + ml) * (long)Kk + 16 * half;

  for (int k0 = 0; k0 < Kk; k0 += 32) {
    float4 a0 = *(const float4*)(arow + k0 + 0);
    float4 a1 = *(const float4*)(arow + k0 + 4);
    float4 a2 = *(const float4*)(arow + k0 + 8);
    float4 a3 = *(const float4*)(arow + k0 + 12);
    const v16h* bp = (const v16h*)(Bf + ((((long)(k0 >> 5)) * mt16 + (tileM >> 4)) * 32 + lane) * 16);
    v16h bf0 = bp[0], bf1 = bp[32], bf2 = bp[64], bf3 = bp[96];   // +1KB per col-tile

    v16h fa;
    fa[0]  = (_Float16)a0.x; fa[1]  = (_Float16)a0.y; fa[2]  = (_Float16)a0.z; fa[3]  = (_Float16)a0.w;
    fa[4]  = (_Float16)a1.x; fa[5]  = (_Float16)a1.y; fa[6]  = (_Float16)a1.z; fa[7]  = (_Float16)a1.w;
    fa[8]  = (_Float16)a2.x; fa[9]  = (_Float16)a2.y; fa[10] = (_Float16)a2.z; fa[11] = (_Float16)a2.w;
    fa[12] = (_Float16)a3.x; fa[13] = (_Float16)a3.y; fa[14] = (_Float16)a3.z; fa[15] = (_Float16)a3.w;
    if (k0 + 32 < Kk) __builtin_prefetch(arow + k0 + 32, 0, 1);

    acc[0] = wmma_f16acc(fa, bf0, acc[0]);
    acc[1] = wmma_f16acc(fa, bf1, acc[1]);
    acc[2] = wmma_f16acc(fa, bf2, acc[2]);
    acc[3] = wmma_f16acc(fa, bf3, acc[3]);
  }

  #pragma unroll
  for (int j = 0; j < 4; ++j) {
    long col = tileM + j * 16 + ml;
    float bv = bias ? bias[col] : 0.0f;          // one load per col-tile
    #pragma unroll
    for (int r = 0; r < 8; ++r) {
      long row = tileN + wave * 16 + r + 8 * half;
      C[row * Mm + col] = acc[j][r] + bv;
    }
  }
}

// =====================================================================
// PASS 1 (parallel over b,h,chunk): local chunk state  CS = Kf^T @ V
// (64x64) and kf column sums CK (64).  grid = NBH*NCH, block = 128.
// =====================================================================
__global__ __launch_bounds__(128)
void attn_state_kernel(const float* __restrict__ K2, const float* __restrict__ V2,
                       float* __restrict__ CS, float* __restrict__ CK)
{
  __shared__ float Kc[16][64];
  __shared__ float Vc[16][64];
  const int blk  = blockIdx.x;
  const int bh   = blk >> 6, cidx = blk & 63;
  const int b    = bh / Hn, hh = bh % Hn;
  const int kvh  = hh / (Hn / HKVn);
  const int s0   = cidx * 16;
  const int tid  = threadIdx.x;
  const int wave = tid >> 5, lane = tid & 31;
  const int half = lane >> 4, ml = lane & 15;
  const int j0   = wave * 16;

  // async-copy V chunk (16x64 f32 = 256 x 16B) into LDS
  #pragma unroll
  for (int it = 0; it < 2; ++it) {
    unsigned c = (unsigned)tid + it * 128u;
    unsigned r = c >> 4, q4 = (c & 15u) * 4u;
    async_lds_b128(V2 + (long)(b * Sn + s0 + r) * KVD + kvh * HDn + q4,
                   (unsigned)(uintptr_t)(&Vc[r][q4]));
  }
  for (int i = tid; i < 16 * 64; i += 128) {
    int r = i >> 6, c = i & 63;
    Kc[r][c] = featmap(K2[(long)(b * Sn + s0 + r) * KVD + kvh * HDn + c]);
  }
  wait_async0();
  __syncthreads();

  float* cs = CS + ((long)bh * NCH + cidx) * 4096;
  #pragma unroll
  for (int fi = 0; fi < 4; ++fi) {
    v8f c = {};
    #pragma unroll
    for (int kk = 0; kk < 16; kk += 4) {
      v2f a, bm;
      a.x  = Kc[kk + 2 * half][fi * 16 + ml];  a.y  = Kc[kk + 2 * half + 1][fi * 16 + ml];
      bm.x = Vc[kk + 2 * half][j0 + ml];       bm.y = Vc[kk + 2 * half + 1][j0 + ml];
      c = wmma_f32k4(a, bm, c);
    }
    #pragma unroll
    for (int r = 0; r < 8; ++r)
      cs[(fi * 16 + r + 8 * half) * 64 + j0 + ml] = c[r];
  }
  if (tid < 64) {
    float s = 0.0f;
    #pragma unroll
    for (int r = 0; r < 16; ++r) s += Kc[r][tid];
    CK[((long)bh * NCH + cidx) * 64 + tid] = s;
  }
}

// =====================================================================
// PASS 2 (32 blocks): in-place *exclusive* prefix scan of CS/CK over
// chunks.  Each thread owns 32 state elements in registers; no LDS.
// =====================================================================
__global__ __launch_bounds__(128)
void attn_scan_kernel(float* __restrict__ CS, float* __restrict__ CK)
{
  const int bh  = blockIdx.x;
  const int tid = threadIdx.x;
  float run[32];
  #pragma unroll
  for (int i = 0; i < 32; ++i) run[i] = 0.0f;
  float runk = 0.0f;
  float* base = CS + (long)bh * NCH * 4096;
  float* kb   = CK + (long)bh * NCH * 64;
  for (int c = 0; c < NCH; ++c) {
    float* p = base + (long)c * 4096;
    #pragma unroll
    for (int i = 0; i < 32; ++i) {
      int e = tid + i * 128;
      float t = p[e];
      p[e] = run[i];
      run[i] += t;
    }
    if (tid < 64) {
      float t = kb[c * 64 + tid];
      kb[c * 64 + tid] = runk;
      runk += t;
    }
  }
}

// =====================================================================
// PASS 3 (parallel over b,h,chunk): O = Qf @ Prefix + tril(Qf@Kf^T) @ V,
// denominators from prefix kf-sums.  Prefix state async-copied to LDS.
// =====================================================================
__global__ __launch_bounds__(128)
void attn_out_kernel(const float* __restrict__ Q, const float* __restrict__ K2,
                     const float* __restrict__ V2, const float* __restrict__ CS,
                     const float* __restrict__ CK, float* __restrict__ O)
{
  __shared__ float Pst[64][64];   // exclusive-prefix KV state
  __shared__ float Qc[16][64];
  __shared__ float Kc[16][64];
  __shared__ float Vc[16][64];
  __shared__ float Sc[16][16];
  __shared__ float dinv[16];

  const int blk  = blockIdx.x;
  const int bh   = blk >> 6, cidx = blk & 63;
  const int b    = bh / Hn, hh = bh % Hn;
  const int kvh  = hh / (Hn / HKVn);
  const int s0   = cidx * 16;
  const int tid  = threadIdx.x;
  const int wave = tid >> 5, lane = tid & 31;
  const int half = lane >> 4, ml = lane & 15;
  const int j0   = wave * 16;

  const float* cs = CS + ((long)bh * NCH + cidx) * 4096;
  const float* ck = CK + ((long)bh * NCH + cidx) * 64;

  // async-copy prefix state (16KB = 1024 x 16B) and V chunk into LDS
  #pragma unroll
  for (int it = 0; it < 8; ++it) {
    int i = tid + it * 128;
    async_lds_b128(cs + (long)i * 4, (unsigned)(uintptr_t)(&Pst[0][0] + i * 4));
  }
  #pragma unroll
  for (int it = 0; it < 2; ++it) {
    unsigned c = (unsigned)tid + it * 128u;
    unsigned r = c >> 4, q4 = (c & 15u) * 4u;
    async_lds_b128(V2 + (long)(b * Sn + s0 + r) * KVD + kvh * HDn + q4,
                   (unsigned)(uintptr_t)(&Vc[r][q4]));
  }
  for (int i = tid; i < 16 * 64; i += 128) {
    int r = i >> 6, c = i & 63;
    long row = (long)(b * Sn + s0 + r);
    Qc[r][c] = featmap(Q[row * Dn + hh * HDn + c]);
    Kc[r][c] = featmap(K2[row * KVD + kvh * HDn + c]);
  }
  wait_async0();
  __syncthreads();

  // wave 0: masked scores Sc = tril(Qf @ Kf^T)
  if (wave == 0) {
    v8f sc = {};
    #pragma unroll
    for (int kk = 0; kk < 64; kk += 4) {
      v2f a, bm;
      a.x  = Qc[ml][kk + 2 * half];  a.y  = Qc[ml][kk + 2 * half + 1];
      bm.x = Kc[ml][kk + 2 * half];  bm.y = Kc[ml][kk + 2 * half + 1];
      sc = wmma_f32k4(a, bm, sc);
    }
    #pragma unroll
    for (int r = 0; r < 8; ++r) {
      int qi = r + 8 * half;
      Sc[qi][ml] = (ml <= qi) ? sc[r] : 0.0f;
    }
  }
  __syncthreads();

  if (tid < 16) {
    float d = 1e-6f;
    #pragma unroll
    for (int f = 0; f < 64; ++f) d += Qc[tid][f] * ck[f];
    for (int j = 0; j <= tid; ++j) d += Sc[tid][j];
    dinv[tid] = 1.0f / d;
  }
  __syncthreads();

  v8f o = {};
  #pragma unroll
  for (int kk = 0; kk < 64; kk += 4) {          // inter: Qf @ Prefix
    v2f a, bm;
    a.x  = Qc[ml][kk + 2 * half];        a.y  = Qc[ml][kk + 2 * half + 1];
    bm.x = Pst[kk + 2 * half][j0 + ml];  bm.y = Pst[kk + 2 * half + 1][j0 + ml];
    o = wmma_f32k4(a, bm, o);
  }
  #pragma unroll
  for (int kk = 0; kk < 16; kk += 4) {          // intra: Sc @ Vc
    v2f a, bm;
    a.x  = Sc[ml][kk + 2 * half];        a.y  = Sc[ml][kk + 2 * half + 1];
    bm.x = Vc[kk + 2 * half][j0 + ml];   bm.y = Vc[kk + 2 * half + 1][j0 + ml];
    o = wmma_f32k4(a, bm, o);
  }
  #pragma unroll
  for (int r = 0; r < 8; ++r) {
    int qi = r + 8 * half;
    long row = (long)(b * Sn + s0 + qi);
    O[row * Dn + hh * HDn + j0 + ml] = o[r] * dinv[qi];
  }
}

// =====================================================================
// Sequential fallback (used if workspace too small for the scan path)
// =====================================================================
__global__ __launch_bounds__(128)
void attn_kernel(const float* __restrict__ Q, const float* __restrict__ K2,
                 const float* __restrict__ V2, float* __restrict__ O)
{
  __shared__ float Sst[64][64];
  __shared__ float ksum[64];
  __shared__ float Qc[16][64];
  __shared__ float Kc[16][64];
  __shared__ float Vc[16][64];
  __shared__ float Sc[16][16];
  __shared__ float dinv[16];

  const int bh  = blockIdx.x;
  const int b   = bh / Hn;
  const int hh  = bh % Hn;
  const int kvh = hh / (Hn / HKVn);
  const int tid  = threadIdx.x;
  const int wave = tid >> 5;
  const int lane = tid & 31;
  const int half = lane >> 4;
  const int ml   = lane & 15;
  const int j0   = wave * 16;

  for (int i = tid; i < 64 * 64; i += 128) (&Sst[0][0])[i] = 0.0f;
  if (tid < 64) ksum[tid] = 0.0f;
  __syncthreads();

  for (int s0 = 0; s0 < Sn; s0 += 16) {
    #pragma unroll
    for (int it = 0; it < 2; ++it) {
      unsigned c = (unsigned)tid + it * 128u;
      unsigned r = c >> 4, q4 = (c & 15u) * 4u;
      async_lds_b128(V2 + (long)(b * Sn + s0 + r) * KVD + kvh * HDn + q4,
                     (unsigned)(uintptr_t)(&Vc[r][q4]));
    }
    for (int i = tid; i < 16 * 64; i += 128) {
      int r = i >> 6, c = i & 63;
      long row = (long)(b * Sn + s0 + r);
      Qc[r][c] = featmap(Q[row * Dn + hh * HDn + c]);
      Kc[r][c] = featmap(K2[row * KVD + kvh * HDn + c]);
    }
    wait_async0();
    __syncthreads();

    if (wave == 0) {
      v8f sc = {};
      #pragma unroll
      for (int kk = 0; kk < 64; kk += 4) {
        v2f a, bm;
        a.x  = Qc[ml][kk + 2 * half];  a.y  = Qc[ml][kk + 2 * half + 1];
        bm.x = Kc[ml][kk + 2 * half];  bm.y = Kc[ml][kk + 2 * half + 1];
        sc = wmma_f32k4(a, bm, sc);
      }
      #pragma unroll
      for (int r = 0; r < 8; ++r) {
        int qi = r + 8 * half;
        Sc[qi][ml] = (ml <= qi) ? sc[r] : 0.0f;
      }
    }
    __syncthreads();

    if (tid < 16) {
      float d = 1e-6f;
      #pragma unroll
      for (int f = 0; f < 64; ++f) d += Qc[tid][f] * ksum[f];
      for (int j = 0; j <= tid; ++j) d += Sc[tid][j];
      dinv[tid] = 1.0f / d;
    }
    __syncthreads();

    {
      v8f o = {};
      #pragma unroll
      for (int kk = 0; kk < 64; kk += 4) {
        v2f a, bm;
        a.x  = Qc[ml][kk + 2 * half];        a.y  = Qc[ml][kk + 2 * half + 1];
        bm.x = Sst[kk + 2 * half][j0 + ml];  bm.y = Sst[kk + 2 * half + 1][j0 + ml];
        o = wmma_f32k4(a, bm, o);
      }
      #pragma unroll
      for (int kk = 0; kk < 16; kk += 4) {
        v2f a, bm;
        a.x  = Sc[ml][kk + 2 * half];        a.y  = Sc[ml][kk + 2 * half + 1];
        bm.x = Vc[kk + 2 * half][j0 + ml];   bm.y = Vc[kk + 2 * half + 1][j0 + ml];
        o = wmma_f32k4(a, bm, o);
      }
      #pragma unroll
      for (int r = 0; r < 8; ++r) {
        int qi = r + 8 * half;
        long row = (long)(b * Sn + s0 + qi);
        O[row * Dn + hh * HDn + j0 + ml] = o[r] * dinv[qi];
      }
    }

    #pragma unroll
    for (int fi = 0; fi < 4; ++fi) {
      v8f c;
      #pragma unroll
      for (int r = 0; r < 8; ++r) c[r] = Sst[fi * 16 + r + 8 * half][j0 + ml];
      #pragma unroll
      for (int kk = 0; kk < 16; kk += 4) {
        v2f a, bm;
        a.x  = Kc[kk + 2 * half][fi * 16 + ml];  a.y  = Kc[kk + 2 * half + 1][fi * 16 + ml];
        bm.x = Vc[kk + 2 * half][j0 + ml];       bm.y = Vc[kk + 2 * half + 1][j0 + ml];
        c = wmma_f32k4(a, bm, c);
      }
      #pragma unroll
      for (int r = 0; r < 8; ++r) Sst[fi * 16 + r + 8 * half][j0 + ml] = c[r];
    }

    if (tid < 64) {
      float s = ksum[tid];
      #pragma unroll
      for (int r = 0; r < 16; ++r) s += Kc[r][tid];
      ksum[tid] = s;
    }
    __syncthreads();
  }
}

// =====================================================================
// residual + LayerNorm: one block per token row
// =====================================================================
__global__ __launch_bounds__(256)
void resid_ln_kernel(const float* __restrict__ x, const float* __restrict__ yp,
                     const float* __restrict__ gamma, const float* __restrict__ beta,
                     float* __restrict__ out)
{
  __shared__ float red[256];
  __shared__ float red2[256];
  const int row = blockIdx.x;
  const int tid = threadIdx.x;
  const long base = (long)row * Dn;

  float vals[4], s = 0.0f, s2 = 0.0f;
  #pragma unroll
  for (int i = 0; i < 4; ++i) {
    int c = tid + i * 256;
    float v = x[base + c] + yp[base + c];
    vals[i] = v; s += v; s2 += v * v;
  }
  red[tid] = s; red2[tid] = s2;
  __syncthreads();
  for (int off = 128; off > 0; off >>= 1) {
    if (tid < off) { red[tid] += red[tid + off]; red2[tid] += red2[tid + off]; }
    __syncthreads();
  }
  float mu  = red[0] * (1.0f / Dn);
  float var = red2[0] * (1.0f / Dn) - mu * mu;
  float inv = rsqrtf(var + 1e-5f);
  #pragma unroll
  for (int i = 0; i < 4; ++i) {
    int c = tid + i * 256;
    out[base + c] = (vals[i] - mu) * inv * gamma[c] + beta[c];
  }
}

// =====================================================================
extern "C" void kernel_launch(void* const* d_in, const int* in_sizes, int n_in,
                              void* d_out, int out_size, void* d_ws, size_t ws_size,
                              hipStream_t stream)
{
  (void)in_sizes; (void)n_in; (void)out_size;
  const float* x     = (const float*)d_in[0];
  const float* Wq    = (const float*)d_in[1];
  const float* bq    = (const float*)d_in[2];
  const float* Wk    = (const float*)d_in[3];
  const float* bk    = (const float*)d_in[4];
  const float* Wv    = (const float*)d_in[5];
  const float* bv    = (const float*)d_in[6];
  const float* Wc    = (const float*)d_in[7];
  const float* We    = (const float*)d_in[8];
  const float* Wo    = (const float*)d_in[9];
  const float* bo    = (const float*)d_in[10];
  const float* gamma = (const float*)d_in[11];
  const float* beta  = (const float*)d_in[12];
  float* out = (float*)d_out;

  float* ws = (float*)d_ws;
  size_t off = 0;
  float* Qb  = ws + off; off += (size_t)Nrow * Dn;
  float* Kr  = ws + off; off += (size_t)Nrow * KVD;
  float* Vr  = ws + off; off += (size_t)Nrow * KVD;
  float* K2  = ws + off; off += (size_t)Nrow * KVD;
  float* V2  = ws + off; off += (size_t)Nrow * KVD;
  float* Wce = ws + off; off += (size_t)KVD * KVD;
  float* Ob  = ws + off; off += (size_t)Nrow * Dn;
  float* Yb  = ws + off; off += (size_t)Nrow * Dn;
  // f16 fragment-ordered weight streams
  _Float16* h16 = (_Float16*)(ws + off);
  size_t hoff = 0;
  _Float16* WqF  = h16 + hoff; hoff += (size_t)Dn * Dn;
  _Float16* WkF  = h16 + hoff; hoff += (size_t)Dn * KVD;
  _Float16* WvF  = h16 + hoff; hoff += (size_t)Dn * KVD;
  _Float16* WeF  = h16 + hoff; hoff += (size_t)64 * KVD;
  _Float16* WceF = h16 + hoff; hoff += (size_t)KVD * KVD;
  _Float16* WoF  = h16 + hoff; hoff += (size_t)Dn * Dn;
  off += (hoff + 1) / 2;
  // chunk-scan buffers (parallel attention path)
  float* CS = ws + off; off += (size_t)NBH * NCH * 64 * 64;   // 32MB
  float* CK = ws + off; off += (size_t)NBH * NCH * 64;
  const bool parallel_attn = (ws_size >= off * sizeof(float));

  dim3 blk(128);
  // ---- pre-swizzle weights into f16 fragment streams ----
  swizzle_w_kernel<<<dim3((Dn/32)*(Dn/16)*32/256),   dim3(256), 0, stream>>>(Wq, WqF, Dn, Dn);
  swizzle_w_kernel<<<dim3((Dn/32)*(KVD/16)*32/256),  dim3(256), 0, stream>>>(Wk, WkF, Dn, KVD);
  swizzle_w_kernel<<<dim3((Dn/32)*(KVD/16)*32/256),  dim3(256), 0, stream>>>(Wv, WvF, Dn, KVD);
  swizzle_w_kernel<<<dim3((64/32)*(KVD/16)*32/256),  dim3(256), 0, stream>>>(We, WeF, 64, KVD);
  swizzle_w_kernel<<<dim3((Dn/32)*(Dn/16)*32/256),   dim3(256), 0, stream>>>(Wo, WoF, Dn, Dn);
  // ---- projections ----
  gemm_wmma_kernel<<<dim3(Dn / 64,  Nrow / 64), blk, 0, stream>>>(x,  WqF, bq, Qb, Nrow, Dn, Dn);
  gemm_wmma_kernel<<<dim3(KVD / 64, Nrow / 64), blk, 0, stream>>>(x,  WkF, bk, Kr, Nrow, Dn, KVD);
  gemm_wmma_kernel<<<dim3(KVD / 64, Nrow / 64), blk, 0, stream>>>(x,  WvF, bv, Vr, Nrow, Dn, KVD);
  // ---- fused latent compress->expand ----
  gemm_wmma_kernel<<<dim3(KVD / 64, KVD / 64),  blk, 0, stream>>>(Wc, WeF, nullptr, Wce, KVD, 64, KVD);
  swizzle_w_kernel<<<dim3((KVD/32)*(KVD/16)*32/256), dim3(256), 0, stream>>>(Wce, WceF, KVD, KVD);
  gemm_wmma_kernel<<<dim3(KVD / 64, Nrow / 64), blk, 0, stream>>>(Kr, WceF, nullptr, K2, Nrow, KVD, KVD);
  gemm_wmma_kernel<<<dim3(KVD / 64, Nrow / 64), blk, 0, stream>>>(Vr, WceF, nullptr, V2, Nrow, KVD, KVD);
  // ---- causal linear attention ----
  if (parallel_attn) {
    attn_state_kernel<<<dim3(NBH * NCH), blk, 0, stream>>>(K2, V2, CS, CK);
    attn_scan_kernel <<<dim3(NBH),       blk, 0, stream>>>(CS, CK);
    attn_out_kernel  <<<dim3(NBH * NCH), blk, 0, stream>>>(Qb, K2, V2, CS, CK, Ob);
  } else {
    attn_kernel<<<dim3(NBH), blk, 0, stream>>>(Qb, K2, V2, Ob);
  }
  // ---- output projection ----
  gemm_wmma_kernel<<<dim3(Dn / 64, Nrow / 64), blk, 0, stream>>>(Ob, WoF, bo, Yb, Nrow, Dn, Dn);
  // ---- residual + layernorm ----
  resid_ln_kernel<<<dim3(Nrow), dim3(256), 0, stream>>>(x, Yb, gamma, beta, out);
}
